// MSFFM_AttentionModule_28638841930386
// MI455X (gfx1250) — compile-verified, matrix operand order checked by reading, not run on hardware
//
#include <hip/hip_runtime.h>
#include <hip/hip_bf16.h>
#include <math.h>

// ---------------------------------------------------------------------------
// Problem constants (from the reference): B=2, C=64, H=W=96 -> N=9216
// O = Bf + FlashAttention(Q=Bf_proj, K=P_proj, V=Bf_proj)
// ---------------------------------------------------------------------------
#define NPIX 9216
#define CH   64
#define BATCH 2
#define KT   64       // keys per chunk
#define WAVES 4       // waves per block (wave32)
#define MT   2        // 16-query m-tiles per wave
#define BLOCK (WAVES * 32)

typedef __attribute__((ext_vector_type(16))) _Float16 v16h;
typedef __attribute__((ext_vector_type(8)))  _Float16 v8h;
typedef __attribute__((ext_vector_type(8)))  float    v8f;

union HFrag { v16h v; v8h h[2]; };

__device__ __forceinline__ v8f wmma_f16(const HFrag& a, const HFrag& b, v8f c) {
  // D = A(16x32 f16) * B(32x16 f16) + C(16x16 f32)
  return __builtin_amdgcn_wmma_f32_16x16x32_f16(
      /*neg_a=*/false, a.v, /*neg_b=*/false, b.v,
      /*c_mod=*/(short)0, c, /*reuse_a=*/false, /*reuse_b=*/false);
}

// 16-lane xor-tree max reduction in the VALU via DPP16 (no LDS traffic).
// quad_perm [1,0,3,2]=0xB1 (xor1), [2,3,0,1]=0x4E (xor2),
// ROW_HALF_MIRROR=0x141 (xor4 once quads uniform), ROW_MIRROR=0x140 (xor8).
__device__ __forceinline__ float dpp_mv(float x, int ctrl_tag) {
  union { float f; int i; } a, b;
  a.f = x;
  switch (ctrl_tag) {
    case 0: b.i = __builtin_amdgcn_update_dpp(0, a.i, 0xB1,  0xF, 0xF, true); break;
    case 1: b.i = __builtin_amdgcn_update_dpp(0, a.i, 0x4E,  0xF, 0xF, true); break;
    case 2: b.i = __builtin_amdgcn_update_dpp(0, a.i, 0x141, 0xF, 0xF, true); break;
    default:b.i = __builtin_amdgcn_update_dpp(0, a.i, 0x140, 0xF, 0xF, true); break;
  }
  return b.f;
}
__device__ __forceinline__ float dpp_max16(float x) {
  x = fmaxf(x, dpp_mv(x, 0));
  x = fmaxf(x, dpp_mv(x, 1));
  x = fmaxf(x, dpp_mv(x, 2));
  x = fmaxf(x, dpp_mv(x, 3));
  return x;
}

// ---------------------------------------------------------------------------
// Stage 1: 1x1-conv projections.
//   Kt[b][n][c] = f16( w_low  @ low  + b_low )   (attention K)
//   Qt[b][n][c] = f16( w_high @ high + b_high )  (attention Q and V)
//   out[b][c][n] = f32( w_high @ high + b_high ) (Bf, residual term)
// ---------------------------------------------------------------------------
__global__ __launch_bounds__(256) void proj_kernel(
    const float* __restrict__ high, const float* __restrict__ low,
    const float* __restrict__ w_low, const float* __restrict__ b_low,
    const float* __restrict__ w_high, const float* __restrict__ b_high,
    _Float16* __restrict__ Kt, _Float16* __restrict__ Qt,
    float* __restrict__ out) {
  __shared__ float wl[CH * CH];
  __shared__ float wh[CH * CH];
  __shared__ float bl[CH];
  __shared__ float bh[CH];

  const int tid = threadIdx.x;
  for (int i = tid; i < CH * CH; i += 256) { wl[i] = w_low[i]; wh[i] = w_high[i]; }
  if (tid < CH) { bl[tid] = b_low[tid]; bh[tid] = b_high[tid]; }
  __syncthreads();

  const int p = blockIdx.x * 256 + tid;   // 0 .. B*N-1
  const int b = p / NPIX;
  const int n = p % NPIX;

  float accL[CH], accH[CH];
#pragma unroll
  for (int o = 0; o < CH; ++o) { accL[o] = bl[o]; accH[o] = bh[o]; }

  for (int c = 0; c < CH; ++c) {
    const float xl = low [((size_t)b * CH + c) * NPIX + n];
    const float xh = high[((size_t)b * CH + c) * NPIX + n];
#pragma unroll
    for (int o = 0; o < CH; ++o) {
      accL[o] = fmaf(wl[o * CH + c], xl, accL[o]);
      accH[o] = fmaf(wh[o * CH + c], xh, accH[o]);
    }
  }

  _Float16* kt = Kt + (size_t)p * CH;
  _Float16* qt = Qt + (size_t)p * CH;
#pragma unroll
  for (int o = 0; o < CH; ++o) {
    kt[o] = (_Float16)accL[o];
    qt[o] = (_Float16)accH[o];
    out[((size_t)b * CH + o) * NPIX + n] = accH[o];
  }
}

// ---------------------------------------------------------------------------
// Stage 2: flash attention (wave32 WMMA).
// Each wave: MT=2 adjacent 16-query tiles (K/V B-fragments reused across both).
// A-fragment layout (16-bit A 16x32): lane L -> row M=L&15, khalf=(L>>4)*8:
//   v16h elems 0..7  = K-dim khalf..+7 ; elems 8..15 = 16+khalf..+7
// C/D layout: v8f elem r -> row M = r + 8*(L>=16), col N = L&15.
// Row sums of the probability tile are computed ON THE MATRIX PIPE via a
// ones-column B-fragment (Osum = P x ones), so the VALU only does the max
// reduction + exp.
// ---------------------------------------------------------------------------
__global__ __launch_bounds__(BLOCK) void attn_kernel(
    const _Float16* __restrict__ Kt, const _Float16* __restrict__ Qt,
    float* __restrict__ out) {
  __shared__ _Float16 ldsK[KT][72];               // [key][c]  (pad 64->72)
  __shared__ _Float16 ldsV[CH][72];               // [c][key]  (transposed V)
  __shared__ _Float16 ldsP[WAVES][MT * 16][72];   // probabilities D->A relayout

  const int qPerBlock = WAVES * MT * 16;                // 128
  const int blocksPerBatch = NPIX / qPerBlock;          // 72
  const int b      = blockIdx.x / blocksPerBatch;
  const int qblk   = blockIdx.x % blocksPerBatch;
  const int tid    = threadIdx.x;
  const int wave   = tid >> 5;
  const int lane   = tid & 31;
  const int lane16 = lane & 15;
  const int khalf  = (lane >> 4) * 8;                   // 0 or 8

  const size_t baseBN = (size_t)b * NPIX * CH;
  const _Float16* Qb = Qt + baseBN;
  const _Float16* Kb = Kt + baseBN;

  const int q0 = qblk * qPerBlock + wave * (MT * 16);
  const int g  = khalf >> 3;                            // 0 or 1

  // Persistent Q A-fragments for both m-tiles.
  HFrag aq[MT][2];
#pragma unroll
  for (int t = 0; t < MT; ++t) {
    const v8h* qrow = (const v8h*)(Qb + (size_t)(q0 + t * 16 + lane16) * CH);
    aq[t][0].h[0] = qrow[g];      aq[t][0].h[1] = qrow[2 + g];  // c 0..31
    aq[t][1].h[0] = qrow[4 + g];  aq[t][1].h[1] = qrow[6 + g];  // c 32..63
  }

  // Ones-column B-fragment: column 0 of a 16-wide tile is all 1.0.
  HFrag bOne;
  {
    const _Float16 ov = (lane16 == 0) ? (_Float16)1.0f : (_Float16)0.0f;
#pragma unroll
    for (int e = 0; e < 16; ++e) bOne.v[e] = ov;
  }

  float run_max[MT][8];
  v8f Oacc[MT][4] = {};
  v8f Osum[MT] = {};                 // col 0 = running softmax row sums
#pragma unroll
  for (int t = 0; t < MT; ++t)
#pragma unroll
    for (int r = 0; r < 8; ++r) run_max[t][r] = -1e30f;

  for (int kb = 0; kb < NPIX; kb += KT) {
    __syncthreads();
    // Cooperative tile fill: K as [key][c], V transposed to [c][key].
#pragma unroll
    for (int it = 0; it < 4; ++it) {
      const int chunk = tid + it * BLOCK;   // 0..511
      const int row   = chunk >> 3;         // 0..63
      const int col8  = (chunk & 7) << 3;   // 0,8,..,56
      const v8h kd = *(const v8h*)(Kb + (size_t)(kb + row) * CH + col8);
      *(v8h*)(&ldsK[row][col8]) = kd;
      const v8h vd = *(const v8h*)(Qb + (size_t)(kb + row) * CH + col8);
#pragma unroll
      for (int e = 0; e < 8; ++e) ldsV[col8 + e][row] = vd[e];
    }
    if (kb + KT < NPIX) {                   // global_prefetch next chunk
      __builtin_prefetch(Kb + (size_t)(kb + KT + (tid >> 1)) * CH, 0, 1);
      __builtin_prefetch(Qb + (size_t)(kb + KT + (tid >> 1)) * CH, 0, 1);
    }
    __syncthreads();

    // Scores: S[t][j] = Q_t^T K_j ; K B-fragments shared by both m-tiles.
    v8f S[MT][4];
#pragma unroll
    for (int j = 0; j < 4; ++j) {
      HFrag bk0, bk1;
      const v8h* krow = (const v8h*)(&ldsK[j * 16 + lane16][0]);
      bk0.h[0] = krow[g];     bk0.h[1] = krow[2 + g];
      bk1.h[0] = krow[4 + g]; bk1.h[1] = krow[6 + g];
#pragma unroll
      for (int t = 0; t < MT; ++t) {
        v8f s = {};
        s = wmma_f16(aq[t][0], bk0, s);
        s = wmma_f16(aq[t][1], bk1, s);
        S[t][j] = s;
      }
    }

    // Online softmax (per m-tile): DPP max reduction, exp, P -> LDS.
#pragma unroll
    for (int t = 0; t < MT; ++t) {
      float alpha[8];
#pragma unroll
      for (int r = 0; r < 8; ++r) {
        float cm = fmaxf(fmaxf(S[t][0][r], S[t][1][r]),
                         fmaxf(S[t][2][r], S[t][3][r]));
        cm = dpp_max16(cm);
        const float nm = fmaxf(run_max[t][r], cm);
        alpha[r] = __expf(run_max[t][r] - nm);
        run_max[t][r] = nm;
      }
#pragma unroll
      for (int cg = 0; cg < 4; ++cg)
#pragma unroll
        for (int r = 0; r < 8; ++r) Oacc[t][cg][r] *= alpha[r];
#pragma unroll
      for (int r = 0; r < 8; ++r) Osum[t][r] *= alpha[r];

#pragma unroll
      for (int r = 0; r < 8; ++r) {
        const int mrow = r + khalf;         // D-layout row M = r + 8*half
#pragma unroll
        for (int j = 0; j < 4; ++j) {
          const float pr = __expf(S[t][j][r] - run_max[t][r]);
          ldsP[wave][t * 16 + mrow][j * 16 + lane16] = (_Float16)pr;
        }
      }
    }
    asm volatile("s_wait_dscnt 0" ::: "memory");  // P writes -> A2 reads

    // O += P * V and Osum += P * ones : V B-fragments shared by both m-tiles.
    HFrag a2[MT][2];
#pragma unroll
    for (int t = 0; t < MT; ++t) {
      const v8h* prow = (const v8h*)(&ldsP[wave][t * 16 + lane16][0]);
      a2[t][0].h[0] = prow[g];     a2[t][0].h[1] = prow[2 + g];   // n 0..31
      a2[t][1].h[0] = prow[4 + g]; a2[t][1].h[1] = prow[6 + g];   // n 32..63
    }
#pragma unroll
    for (int cg = 0; cg < 4; ++cg) {
      HFrag bv0, bv1;
      const v8h* vrow = (const v8h*)(&ldsV[cg * 16 + lane16][0]);
      bv0.h[0] = vrow[g];     bv0.h[1] = vrow[2 + g];
      bv1.h[0] = vrow[4 + g]; bv1.h[1] = vrow[6 + g];
#pragma unroll
      for (int t = 0; t < MT; ++t) {
        Oacc[t][cg] = wmma_f16(a2[t][0], bv0, Oacc[t][cg]);
        Oacc[t][cg] = wmma_f16(a2[t][1], bv1, Oacc[t][cg]);
      }
    }
#pragma unroll
    for (int t = 0; t < MT; ++t) {          // row sums on the matrix pipe
      Osum[t] = wmma_f16(a2[t][0], bOne, Osum[t]);
      Osum[t] = wmma_f16(a2[t][1], bOne, Osum[t]);
    }
  }

  // Epilogue: out (already holds Bf) += O / row_sum.
  // Row sum of row (r + 8*half) sits in column 0 -> lanes 0 / 16.
#pragma unroll
  for (int t = 0; t < MT; ++t) {
    float inv[8];
#pragma unroll
    for (int r = 0; r < 8; ++r) {
      const float s = __shfl(Osum[t][r], lane & 16, 32);
      inv[r] = 1.0f / s;
    }
#pragma unroll
    for (int cg = 0; cg < 4; ++cg) {
      const int c = cg * 16 + lane16;
#pragma unroll
      for (int r = 0; r < 8; ++r) {
        const int q = q0 + t * 16 + r + khalf;
        const size_t idx = ((size_t)b * CH + c) * NPIX + q;
        out[idx] += Oacc[t][cg][r] * inv[r];
      }
    }
  }
}

// ---------------------------------------------------------------------------
extern "C" void kernel_launch(void* const* d_in, const int* in_sizes, int n_in,
                              void* d_out, int out_size, void* d_ws, size_t ws_size,
                              hipStream_t stream) {
  (void)in_sizes; (void)n_in; (void)out_size; (void)ws_size;
  const float* high   = (const float*)d_in[0];
  const float* low    = (const float*)d_in[1];
  const float* w_low  = (const float*)d_in[2];
  const float* b_low  = (const float*)d_in[3];
  const float* w_high = (const float*)d_in[4];
  const float* b_high = (const float*)d_in[5];
  float* out = (float*)d_out;

  _Float16* Kt = (_Float16*)d_ws;                        // [B][N][C] f16
  _Float16* Qt = Kt + (size_t)BATCH * NPIX * CH;         // [B][N][C] f16

  proj_kernel<<<dim3((BATCH * NPIX) / 256), dim3(256), 0, stream>>>(
      high, low, w_low, b_low, w_high, b_high, Kt, Qt, out);

  attn_kernel<<<dim3(BATCH * (NPIX / (WAVES * MT * 16))), dim3(BLOCK), 0, stream>>>(
      Kt, Qt, out);
}
